// GraphDecoder_2104533975517
// MI455X (gfx1250) — matrix operand order repaired and checked by reading, not compiled, not run on hardware
//
#include <hip/hip_runtime.h>
#include <hip/hip_bf16.h>
#include <math.h>

typedef __bf16 bf16;
typedef __attribute__((ext_vector_type(16))) __bf16 v16bf;
typedef __attribute__((ext_vector_type(8)))  __bf16 v8bf;
typedef __attribute__((ext_vector_type(8)))  float  v8f;

#define NB   128
#define NN   1000
#define EE   128
#define HH   8
#define CEc  384
#define HD   48
#define NT   63
#define NPAD 1008

// ---- WMMA fragment helpers (contiguous k order k = 16h + s, same for A and B) ----

// A 16x32 bf16: lanes 0-15 row M=lane hold K = kc+0..15 (elems 0..15);
// lanes 16-31 row M=lane-16 hold K = kc+16..31.
__device__ __forceinline__ v16bf load_a_frag(const bf16* abuf, int lane, int kc) {
  int row = lane & 15;
  int hi  = lane >> 4;
  const bf16* rp = abuf + row * EE + kc + hi * 16;
  v8bf lo  = *(const v8bf*)(rp);
  v8bf hi8 = *(const v8bf*)(rp + 8);
  v16bf a;
#pragma unroll
  for (int j = 0; j < 8; ++j) { a[j] = lo[j]; a[j + 8] = hi8[j]; }
  return a;
}

// B 32x16 bf16 of W^T: column N = row of W. lanes 0-15 hold K=kc..kc+15,
// lanes 16-31 hold K=kc+16..kc+31; 16 consecutive bf16 from W[nrow].
__device__ __forceinline__ v16bf load_b_frag(const bf16* W, int ldw, int nrow,
                                             int lane, int kc) {
  int hi = lane >> 4;
  const bf16* rp = W + (size_t)nrow * ldw + kc + hi * 16;
  v8bf b0 = *(const v8bf*)(rp);
  v8bf b1 = *(const v8bf*)(rp + 8);
  v16bf b;
#pragma unroll
  for (int j = 0; j < 8; ++j) { b[j] = b0[j]; b[j + 8] = b1[j]; }
  return b;
}

__device__ __forceinline__ v8f wmma_bf16(v16bf a, v16bf b, v8f c) {
  return __builtin_amdgcn_wmma_f32_16x16x32_bf16(false, a, false, b,
                                                 (short)0, c, false, false);
}

// ---- weight bf16 conversion ----
__global__ void prep_weights(const float* __restrict__ kw,
                             const float* __restrict__ vw,
                             const float* __restrict__ kpw,
                             bf16* __restrict__ kwb,
                             bf16* __restrict__ vwb,
                             bf16* __restrict__ kpb) {
  int i = blockIdx.x * blockDim.x + threadIdx.x;
  if (i < CEc * EE) {
    kwb[i] = (bf16)kw[i];
    vwb[i] = (bf16)vw[i];
  }
  if (i < EE * EE) kpb[i] = (bf16)kpw[i];
}

// ---- fused per-batch decoder ----
__global__ __launch_bounds__(384)
void decoder_fused(const float* __restrict__ node_embs,
                   const int*   __restrict__ mask,
                   const float* __restrict__ gumbel,
                   const float* __restrict__ first_node,
                   const float* __restrict__ last_node,
                   const float* __restrict__ q_w,
                   const float* __restrict__ in_b,
                   const float* __restrict__ out_w,
                   const float* __restrict__ out_b,
                   const float* __restrict__ att_w,
                   const int*   __restrict__ Cptr,
                   const bf16*  __restrict__ kwb,
                   const bf16*  __restrict__ vwb,
                   const bf16*  __restrict__ kpb,
                   float* __restrict__ out) {
  __shared__ __align__(16) unsigned char smem[61440];
  float* kt   = (float*)smem;             // 16*384 f32 K tile
  float* vt   = (float*)(smem + 24576);   // 16*384 f32 V tile (phase-1)
  bf16*  abuf = (bf16*)(smem + 49152);    // 16*128 bf16 A staging
  float* misc = (float*)(smem + 53248);   // 2048 f32
  float* qfull = misc;                    // 384
  float* ctxv  = misc + 384;              // 384 (context / ctx)
  float* ctxO  = misc + 768;              // 384 (temp partials / out-proj)
  float* qp    = misc + 1152;             // 128
  float* sS    = misc + 1280;             // 128 scores
  float* pS    = misc + 1408;             // 128 probs
  float* mh    = misc + 1536;             // 8
  float* lh    = misc + 1544;             // 8
  float* sch   = misc + 1552;             // 8
  // phase-2 overlays inside vt region (vt is dead after attention loop)
  float* kpt   = vt;                      // 16*128 kp tile
  float* redf  = vt + 2048;               // 384
  int*   redi  = (int*)(vt + 2432);       // 384
  float* u_all = vt + 3072;               // 1008
  float* part  = vt + 4096;               // 256

  const int t    = threadIdx.x;
  const int b    = blockIdx.x;
  const int wave = t >> 5;
  const int lane = t & 31;
  const float NEG = -INFINITY;
  const float* nb_base = node_embs + (size_t)b * NN * EE;

  // 1) graph mean + context
  {
    int g = t >> 7, e = t & 127;
    float ps = 0.f;
    for (int n = g; n < NN; n += 3) ps += nb_base[(size_t)n * EE + e];
    ctxO[g * EE + e] = ps;
  }
  __syncthreads();
  if (t < EE)            ctxv[t] = (ctxO[t] + ctxO[EE + t] + ctxO[2 * EE + t]) * (1.0f / NN);
  else if (t < 2 * EE)   ctxv[t] = first_node[t - EE];
  else                   ctxv[t] = last_node[t - 2 * EE];
  if (t < 8) { mh[t] = NEG; lh[t] = 0.f; sch[t] = 0.f; }
  __syncthreads();

  // 2) q = context @ q_w^T + bq
  {
    float s = in_b[t];
    const float* row = q_w + (size_t)t * CEc;
    for (int j = 0; j < CEc; ++j) s += ctxv[j] * row[j];
    qfull[t] = s;
  }
  __syncthreads();

  float acc = 0.f;              // ctx accumulator for channel t
  const int hOwn = t / HD;      // head owning channel t

  // 3) online-softmax attention over 63 node tiles; K/V via WMMA
  for (int tile = 0; tile < NT; ++tile) {
    const int n0 = tile * 16;
    for (int idx = t; idx < 16 * EE; idx += 384) {
      int r = idx >> 7, e = idx & 127;
      int n = n0 + r;
      float v = (n < NN) ? nb_base[(size_t)n * EE + e] : 0.f;
      abuf[idx] = (bf16)v;
      // prefetch the same element of the next tile (emits global_prefetch)
      if (n + 16 < NN)
        __builtin_prefetch(nb_base + (size_t)(n + 16) * EE + e, 0, 3);
    }
    __syncthreads();
    // A fragments are job-invariant: load once per tile into registers
    {
      v16bf a0 = load_a_frag(abuf, lane, 0);
      v16bf a1 = load_a_frag(abuf, lane, 32);
      v16bf a2 = load_a_frag(abuf, lane, 64);
      v16bf a3 = load_a_frag(abuf, lane, 96);
      // 48 tile-jobs (24 K-tiles + 24 V-tiles), 4 per wave
      for (int j = wave * 4; j < wave * 4 + 4; ++j) {
        const int  isV   = (j >= 24);
        const int  ntile = isV ? (j - 24) : j;
        const bf16* W    = isV ? vwb : kwb;
        float*     dst   = isV ? vt : kt;
        const float* bias = in_b + (isV ? 2 * CEc : CEc);
        const int nrow = ntile * 16 + (lane & 15);
        v8f c = {0.f, 0.f, 0.f, 0.f, 0.f, 0.f, 0.f, 0.f};
        c = wmma_bf16(a0, load_b_frag(W, EE, nrow, lane, 0),  c);
        c = wmma_bf16(a1, load_b_frag(W, EE, nrow, lane, 32), c);
        c = wmma_bf16(a2, load_b_frag(W, EE, nrow, lane, 64), c);
        c = wmma_bf16(a3, load_b_frag(W, EE, nrow, lane, 96), c);
        int ch = nrow;
        int hi = lane >> 4;
        float bv = bias[ch];
#pragma unroll
        for (int r = 0; r < 8; ++r) dst[(r + hi * 8) * CEc + ch] = c[r] + bv;
      }
    }
    __syncthreads();
    // per-head scores (8 heads x 16 nodes)
    if (t < 128) {
      int h = t >> 4, nl = t & 15;
      int n = n0 + nl;
      float sc = NEG;
      if (n < NN && mask[(size_t)b * NN + n] == 0) {
        float d = 0.f;
        const float* qh = qfull + h * HD;
        const float* kr = kt + nl * CEc + h * HD;
#pragma unroll
        for (int dd = 0; dd < HD; ++dd) d += qh[dd] * kr[dd];
        sc = d * 0.14433756729740643f;   // 1/sqrt(48)
      }
      sS[t] = sc;
    }
    __syncthreads();
    if (t < 8) {  // one thread per head updates m/l and probs
      float mo = mh[t];
      float tm = NEG;
#pragma unroll
      for (int n = 0; n < 16; ++n) tm = fmaxf(tm, sS[t * 16 + n]);
      float mn = fmaxf(mo, tm);
      float scl, tsum = 0.f;
      if (mn == NEG) {
        scl = 1.f;
#pragma unroll
        for (int n = 0; n < 16; ++n) pS[t * 16 + n] = 0.f;
      } else {
        scl = (mo == NEG) ? 0.f : __expf(mo - mn);
#pragma unroll
        for (int n = 0; n < 16; ++n) {
          float sv = sS[t * 16 + n];
          float pv = (sv == NEG) ? 0.f : __expf(sv - mn);
          pS[t * 16 + n] = pv;
          tsum += pv;
        }
      }
      mh[t] = mn;
      lh[t] = lh[t] * scl + tsum;
      sch[t] = scl;
    }
    __syncthreads();
    {
      float scl = sch[hOwn];
      float add = 0.f;
      const float* pr = pS + hOwn * 16;
#pragma unroll
      for (int n = 0; n < 16; ++n) add += pr[n] * vt[n * CEc + t];
      acc = acc * scl + add;
    }
    __syncthreads();
  }

  // 4) normalize ctx, out-projection, qp = ctx2 @ att_w^T
  ctxv[t] = acc / lh[hOwn];
  __syncthreads();
  {
    float s = out_b[t];
    const float* row = out_w + (size_t)t * CEc;
    for (int j = 0; j < CEc; ++j) s += ctxv[j] * row[j];
    ctxO[t] = s;
  }
  __syncthreads();
  if (t < EE) {
    float s = 0.f;
    const float* row = att_w + (size_t)t * CEc;
    for (int j = 0; j < CEc; ++j) s += ctxO[j] * row[j];
    qp[t] = s;
  }
  __syncthreads();

  const float Cf = (float)(*Cptr);

  // 5) kp via WMMA + u = tanh(qp.kp/sqrt(e))*C  (masked -> -inf)
  for (int tile = 0; tile < NT; ++tile) {
    const int n0 = tile * 16;
    for (int idx = t; idx < 16 * EE; idx += 384) {
      int r = idx >> 7, e = idx & 127;
      int n = n0 + r;
      float v = (n < NN) ? nb_base[(size_t)n * EE + e] : 0.f;
      abuf[idx] = (bf16)v;
      if (n + 16 < NN)
        __builtin_prefetch(nb_base + (size_t)(n + 16) * EE + e, 0, 3);
    }
    __syncthreads();
    if (wave < 8) {  // wave-uniform branch: EXEC all-ones inside
      v16bf a0 = load_a_frag(abuf, lane, 0);
      v16bf a1 = load_a_frag(abuf, lane, 32);
      v16bf a2 = load_a_frag(abuf, lane, 64);
      v16bf a3 = load_a_frag(abuf, lane, 96);
      const int nrow = wave * 16 + (lane & 15);
      v8f c = {0.f, 0.f, 0.f, 0.f, 0.f, 0.f, 0.f, 0.f};
      c = wmma_bf16(a0, load_b_frag(kpb, EE, nrow, lane, 0),  c);
      c = wmma_bf16(a1, load_b_frag(kpb, EE, nrow, lane, 32), c);
      c = wmma_bf16(a2, load_b_frag(kpb, EE, nrow, lane, 64), c);
      c = wmma_bf16(a3, load_b_frag(kpb, EE, nrow, lane, 96), c);
      int hi = lane >> 4;
#pragma unroll
      for (int r = 0; r < 8; ++r) kpt[(r + hi * 8) * EE + nrow] = c[r];
    }
    __syncthreads();
    if (t < 256) {  // 16 nodes x 16 segments of 8
      int nl = t >> 4, seg = t & 15;
      const float* kr = kpt + nl * EE + seg * 8;
      const float* qr = qp + seg * 8;
      float d = 0.f;
#pragma unroll
      for (int j = 0; j < 8; ++j) d += qr[j] * kr[j];
      part[nl * 16 + seg] = d;
    }
    __syncthreads();
    if (t < 16) {
      int n = n0 + t;
      float uu = NEG;
      if (n < NN && mask[(size_t)b * NN + n] == 0) {
        float d = 0.f;
#pragma unroll
        for (int s2 = 0; s2 < 16; ++s2) d += part[t * 16 + s2];
        uu = tanhf(d * 0.08838834764831845f) * Cf;   // 1/sqrt(128)
      }
      u_all[n0 + t] = uu;
    }
    __syncthreads();
  }

  // 6) argmax(u + gumbel) and logsumexp(u)
  float lm = NEG, bestv = NEG;
  int besti = 0;
  for (int n = t; n < NPAD; n += 384) {
    float u = u_all[n];
    lm = fmaxf(lm, u);
    float g = (n < NN) ? gumbel[(size_t)b * NN + n] : 0.f;
    float w = u + g;
    if (w > bestv || (w == bestv && n < besti)) { bestv = w; besti = n; }
  }
  redf[t] = bestv; redi[t] = besti;
  __syncthreads();
  for (int s = 256; s > 0; s >>= 1) {
    if (t < s && t + s < 384) {
      float o = redf[t + s];
      if (o > redf[t] || (o == redf[t] && redi[t + s] < redi[t])) {
        redf[t] = o; redi[t] = redi[t + s];
      }
    }
    __syncthreads();
  }
  int nn_idx = redi[0];
  __syncthreads();
  redf[t] = lm;
  __syncthreads();
  for (int s = 256; s > 0; s >>= 1) {
    if (t < s && t + s < 384) redf[t] = fmaxf(redf[t], redf[t + s]);
    __syncthreads();
  }
  float um = redf[0];
  __syncthreads();
  float ps = 0.f;
  for (int n = t; n < NPAD; n += 384) ps += __expf(u_all[n] - um);
  redf[t] = ps;
  __syncthreads();
  for (int s = 256; s > 0; s >>= 1) {
    if (t < s && t + s < 384) redf[t] += redf[t + s];
    __syncthreads();
  }
  if (t == 0) {
    float lse = um + __logf(redf[0]);
    out[b]             = (float)nn_idx;            // nn_idx
    out[gridDim.x + b] = u_all[nn_idx] - lse;      // logp
  }
}

extern "C" void kernel_launch(void* const* d_in, const int* in_sizes, int n_in,
                              void* d_out, int out_size, void* d_ws, size_t ws_size,
                              hipStream_t stream) {
  (void)in_sizes; (void)n_in; (void)out_size; (void)ws_size;
  const float* node_embs  = (const float*)d_in[0];
  const int*   mask       = (const int*)d_in[1];
  const float* gumbel     = (const float*)d_in[2];
  const float* first_node = (const float*)d_in[3];
  const float* last_node  = (const float*)d_in[4];
  const float* q_w   = (const float*)d_in[5];
  const float* k_w   = (const float*)d_in[6];
  const float* v_w   = (const float*)d_in[7];
  const float* in_b  = (const float*)d_in[8];
  const float* out_w = (const float*)d_in[9];
  const float* out_b = (const float*)d_in[10];
  const float* kp_w  = (const float*)d_in[11];
  const float* att_w = (const float*)d_in[12];
  const int*   Cptr  = (const int*)d_in[13];

  bf16* kwb = (bf16*)d_ws;            // 384*128 bf16
  bf16* vwb = kwb + CEc * EE;         // 384*128 bf16
  bf16* kpb = vwb + CEc * EE;         // 128*128 bf16

  prep_weights<<<(CEc * EE + 255) / 256, 256, 0, stream>>>(k_w, v_w, kp_w,
                                                           kwb, vwb, kpb);
  decoder_fused<<<NB, 384, 0, stream>>>(node_embs, mask, gumbel, first_node,
                                        last_node, q_w, in_b, out_w, out_b,
                                        att_w, Cptr, kwb, vwb, kpb,
                                        (float*)d_out);
}
